// MultiHeadAttention_81793357185364
// MI455X (gfx1250) — matrix operand order repaired and checked by reading, not compiled, not run on hardware
//
#include <hip/hip_runtime.h>
#include <hip/hip_bf16.h>

// ---------------- problem constants ----------------
constexpr int kB  = 8;
constexpr int kS  = 1024;
constexpr int kD  = 1024;
constexpr int kH  = 16;
constexpr int kDK = 64;

typedef __attribute__((ext_vector_type(16))) __bf16   v16bf;
typedef __attribute__((ext_vector_type(8)))  float    v8f;
typedef __attribute__((ext_vector_type(4)))  float    v4f;
typedef __attribute__((ext_vector_type(4)))  unsigned v4u;
typedef __attribute__((ext_vector_type(2)))  unsigned v2u;

// WMMA fragment (CDNA5 ISA §7.12.2, wave32):
//  A (16x32 bf16): lane l -> row m = l&15; half = l>>4;
//     VGPR v: k = half*16 + 2v (contiguous order used here)  -> two contiguous 16B runs per lane
//  B (32x16 bf16): lane l -> col n = l&15; half = l>>4; VGPR v: k = half*16 + 2v (same order as A, not the ISA table)
//  C/D (16x16 f32): lane l -> col n = l&15; VGPR v -> row m = v + (l>>4)*8
union Frag {
    v16bf    f;
    v4u      q2[2];
    unsigned w[8];
};

__device__ inline unsigned pack_bf16_f32(float lo, float hi) {
    union { unsigned u; __bf16 h[2]; } r;
    r.h[0] = (__bf16)lo;          // -> v_cvt_pk_bf16_f32
    r.h[1] = (__bf16)hi;
    return r.u;
}
__device__ inline unsigned pack2_bf16(__bf16 lo, __bf16 hi) {
    union { unsigned u; __bf16 h[2]; } r;
    r.h[0] = lo; r.h[1] = hi;
    return r.u;
}

#define WMMA_BF16(A, Bm, C) \
    __builtin_amdgcn_wmma_f32_16x16x32_bf16(false, (A), false, (Bm), (short)0, (C), false, false)

// ===================== QKV projection =====================
// Y[b,h,s,n] = sum_k X[b,s,k] * W[h,k,n] + bias[h,n]   (bf16 out)
// 64x64 block, 8 waves: wave = (mtile 0..3) x (nhalf 0..1), 2 WMMAs/wave/K-step.
__global__ __launch_bounds__(256)
void qkv_proj_kernel(const float* __restrict__ X,    // [B,S,D]
                     const float* __restrict__ W,    // [H,D,DK]
                     const float* __restrict__ bias, // [H,DK]
                     __bf16* __restrict__ Y)         // [B,H,S,DK]
{
    __shared__ __bf16   sA[64][40];   // 64 rows x 32 K (+pad); row = 80B (16B aligned)
    __shared__ unsigned sBp[16][68];  // packed pairs: [kpair][n], word = (k, k+1)

    const int tid   = threadIdx.x;
    const int lane  = tid & 31;
    const int wave  = tid >> 5;
    const int n16   = lane & 15;
    const int halfl = lane >> 4;

    const int s0 = blockIdx.x * 64;
    const int h  = blockIdx.y;
    const int b  = blockIdx.z;

    const int mtile = wave >> 1;   // 0..3
    const int nhalf = wave & 1;    // 0..1

    const float* Xbase = X + ((size_t)b * kS + s0) * kD;
    const float* Wbase = W + (size_t)h * kD * kDK;

    // staging coordinates (branch-free, one vector op each)
    const int ar  = tid >> 3;         // 0..31 (row; +32 for second half)
    const int aq  = (tid & 7) * 4;    // float4 column within 32-K slice
    const int br2 = tid >> 4;         // 0..15 k-pair
    const int bc4 = (tid & 15) * 4;   // 4-wide n column

    v8f accs[2] = {{}, {}};

    for (int k0 = 0; k0 < kD; k0 += 32) {
        // ---- stage A: 64 rows x 32 K, fp32 -> bf16, b128 loads / b64 stores ----
        #pragma unroll
        for (int j = 0; j < 2; ++j) {
            const int r = ar + 32 * j;
            v4f x = *(const v4f*)(Xbase + (size_t)r * kD + k0 + aq);
            v2u w;
            w[0] = pack_bf16_f32(x[0], x[1]);
            w[1] = pack_bf16_f32(x[2], x[3]);
            *(v2u*)(&sA[r][aq]) = w;
        }
        // ---- stage B packed pairs: 16 kpairs x 64 n, two b128 loads, b128 store ----
        {
            const float* r0 = Wbase + (size_t)(k0 + 2 * br2) * kDK + bc4;
            v4f wa = *(const v4f*)(r0);
            v4f wb = *(const v4f*)(r0 + kDK);
            v4u w;
            #pragma unroll
            for (int j = 0; j < 4; ++j) w[j] = pack_bf16_f32(wa[j], wb[j]);
            *(v4u*)(&sBp[br2][bc4]) = w;
        }
        // speculative prefetch of next K slice (global_prefetch_b8; OOB dropped by HW)
        __builtin_prefetch(Xbase + (size_t)ar * kD + k0 + 32 + aq, 0, 0);
        __syncthreads();

        // ---- fragments: A = 2x ds_load_b128; B = 8x ds_load_b32 each ----
        Frag a;
        const __bf16* arow = &sA[mtile * 16 + n16][0];
        a.q2[0] = *(const v4u*)(arow + halfl * 16);
        a.q2[1] = *(const v4u*)(arow + halfl * 16 + 8);

        Frag b0, b1;
        #pragma unroll
        for (int v = 0; v < 8; ++v) {
            b0.w[v] = sBp[halfl * 8 + v][(nhalf * 2 + 0) * 16 + n16];
            b1.w[v] = sBp[halfl * 8 + v][(nhalf * 2 + 1) * 16 + n16];
        }
        accs[0] = WMMA_BF16(a.f, b0.f, accs[0]);
        accs[1] = WMMA_BF16(a.f, b1.f, accs[1]);
        __syncthreads();
    }

    // ---- epilogue: bias + bf16 store ----
    __bf16* Yb = Y + (((size_t)b * kH + h) * kS + s0 + mtile * 16) * kDK;
    #pragma unroll
    for (int t = 0; t < 2; ++t) {
        const int ntile = nhalf * 2 + t;
        const float bval = bias[h * kDK + ntile * 16 + n16];
        #pragma unroll
        for (int v = 0; v < 8; ++v) {
            const int m = v + halfl * 8;
            Yb[(size_t)m * kDK + ntile * 16 + n16] = (__bf16)(accs[t][v] + bval);
        }
    }
}

// ===================== attention (scores + softmax + ctx) =====================
__global__ __launch_bounds__(256)
void attn_kernel(const __bf16* __restrict__ Q,   // [B,H,S,DK]
                 const __bf16* __restrict__ K,   // [B,H,S,DK]
                 const __bf16* __restrict__ V,   // [B,H,S,DK]
                 float* __restrict__ attn_out,   // [B,H,S,S]
                 __bf16* __restrict__ ctx_out)   // [B,S,H*DK]
{
    __shared__ __bf16   sP[16][1024];     // normalized probabilities (32 KB)
    __shared__ unsigned sVp[32][68];      // packed V pairs for one 64-K chunk (8.5 KB)
    __shared__ float    red[8][16];
    __shared__ float    rowmax[16];
    __shared__ float    rowinv[16];
    __shared__ float    cpart[2][16][64]; // split-K partial ctx (8 KB)

    const int tid   = threadIdx.x;
    const int lane  = tid & 31;
    const int wave  = tid >> 5;
    const int n16   = lane & 15;
    const int halfl = lane >> 4;

    const int s0 = blockIdx.x * 16;
    const int h  = blockIdx.y;
    const int b  = blockIdx.z;

    const __bf16* Qb = Q + (((size_t)b * kH + h) * kS) * kDK;
    const __bf16* Kb = K + (((size_t)b * kH + h) * kS) * kDK;
    const __bf16* Vb = V + (((size_t)b * kH + h) * kS) * kDK;

    // Q fragments (rows s0..s0+15, DK=64 -> two K=32 fragments), 4x global b128
    Frag qa0, qa1;
    {
        const __bf16* qrow = Qb + (size_t)(s0 + n16) * kDK;
        qa0.q2[0] = *(const v4u*)(qrow + halfl * 16);
        qa0.q2[1] = *(const v4u*)(qrow + halfl * 16 + 8);
        qa1.q2[0] = *(const v4u*)(qrow + 32 + halfl * 16);
        qa1.q2[1] = *(const v4u*)(qrow + 32 + halfl * 16 + 8);
    }

    // ---- phase 1: scores; wave w owns t-tiles (w + 8*tt) ----
    const float scale = 0.125f;  // 1/sqrt(64)
    v8f sc[8];
    #pragma unroll
    for (int tt = 0; tt < 8; ++tt) {
        const int t0 = (wave + tt * 8) * 16;
        const __bf16* krow = Kb + (size_t)(t0 + n16) * kDK;  // B(k,n) = K[t0+n][k]
        Frag kb0, kb1;  // contiguous per lane: 4x global b128
        kb0.q2[0] = *(const v4u*)(krow + halfl * 16);
        kb0.q2[1] = *(const v4u*)(krow + halfl * 16 + 8);
        kb1.q2[0] = *(const v4u*)(krow + 32 + halfl * 16);
        kb1.q2[1] = *(const v4u*)(krow + 32 + halfl * 16 + 8);
        v8f c = {};
        c = WMMA_BF16(qa0.f, kb0.f, c);
        c = WMMA_BF16(qa1.f, kb1.f, c);
        #pragma unroll
        for (int v = 0; v < 8; ++v) c[v] *= scale;
        sc[tt] = c;
    }

    // ---- row max: shuffle reduce within wave, LDS across waves ----
    float lmax[8];
    #pragma unroll
    for (int v = 0; v < 8; ++v) {
        float m = sc[0][v];
        #pragma unroll
        for (int tt = 1; tt < 8; ++tt) m = fmaxf(m, sc[tt][v]);
        #pragma unroll
        for (int off = 1; off < 16; off <<= 1)
            m = fmaxf(m, __shfl_xor(m, off, 32));
        lmax[v] = m;
    }
    if (n16 == 0) {
        #pragma unroll
        for (int v = 0; v < 8; ++v) red[wave][halfl * 8 + v] = lmax[v];
    }
    __syncthreads();
    if (tid < 16) {
        float m = red[0][tid];
        #pragma unroll
        for (int w = 1; w < 8; ++w) m = fmaxf(m, red[w][tid]);
        rowmax[tid] = m;
    }
    __syncthreads();

    // ---- exp + row sum ----
    float lsum[8];
    #pragma unroll
    for (int v = 0; v < 8; ++v) {
        const float rm = rowmax[halfl * 8 + v];
        float s = 0.f;
        #pragma unroll
        for (int tt = 0; tt < 8; ++tt) {
            float p = __expf(sc[tt][v] - rm);
            sc[tt][v] = p;
            s += p;
        }
        #pragma unroll
        for (int off = 1; off < 16; off <<= 1)
            s += __shfl_xor(s, off, 32);
        lsum[v] = s;
    }
    if (n16 == 0) {
        #pragma unroll
        for (int v = 0; v < 8; ++v) red[wave][halfl * 8 + v] = lsum[v];
    }
    __syncthreads();
    if (tid < 16) {
        float s = red[0][tid];
        #pragma unroll
        for (int w = 1; w < 8; ++w) s += red[w][tid];
        rowinv[tid] = 1.0f / s;
    }
    __syncthreads();

    // ---- normalize: write attn (fp32 output) + P (bf16 LDS) ----
    float* attn_b = attn_out + (((size_t)b * kH + h) * kS + s0) * kS;
    #pragma unroll
    for (int tt = 0; tt < 8; ++tt) {
        const int t0 = (wave + tt * 8) * 16;
        #pragma unroll
        for (int v = 0; v < 8; ++v) {
            const int r = halfl * 8 + v;
            const float p = sc[tt][v] * rowinv[r];
            attn_b[(size_t)r * kS + t0 + n16] = p;
            sP[r][t0 + n16] = (__bf16)p;
        }
    }
    __syncthreads();

    // ---- phase 2: ctx = P @ V over 16 chunks of K=64; LDS-staged packed V ----
    const int ntile = wave & 3;
    const int ksub  = wave >> 2;     // split-K half
    const int vr2 = tid >> 3;        // 0..31 k-pair within chunk
    const int vc8 = (tid & 7) * 8;   // 8-wide n column
    v8f cacc = {};
    for (int ch = 0; ch < 16; ++ch) {
        const int ko = ch * 64;
        {   // stage V[ko..ko+63][0..63] as packed pairs (2x global b128 per thread)
            const __bf16* r0 = Vb + (size_t)(ko + 2 * vr2) * kDK + vc8;
            union { v4u q; __bf16 hh[8]; } ra, rb;
            ra.q = *(const v4u*)(r0);
            rb.q = *(const v4u*)(r0 + kDK);
            v4u w0, w1;
            #pragma unroll
            for (int j = 0; j < 4; ++j) {
                w0[j] = pack2_bf16(ra.hh[j],     rb.hh[j]);
                w1[j] = pack2_bf16(ra.hh[4 + j], rb.hh[4 + j]);
            }
            *(v4u*)(&sVp[vr2][vc8])     = w0;
            *(v4u*)(&sVp[vr2][vc8 + 4]) = w1;
        }
        __syncthreads();

        const int k0 = ko + ksub * 32;
        Frag a, bm;
        a.q2[0] = *(const v4u*)(&sP[n16][k0 + halfl * 16]);
        a.q2[1] = *(const v4u*)(&sP[n16][k0 + halfl * 16 + 8]);
        #pragma unroll
        for (int v = 0; v < 8; ++v)
            bm.w[v] = sVp[ksub * 16 + halfl * 8 + v][ntile * 16 + n16];
        cacc = WMMA_BF16(a.f, bm.f, cacc);
        __syncthreads();
    }
    #pragma unroll
    for (int v = 0; v < 8; ++v)
        cpart[ksub][v + halfl * 8][ntile * 16 + n16] = cacc[v];
    __syncthreads();

    if (ksub == 0) {
        __bf16* cb = ctx_out + ((size_t)b * kS + s0) * (kH * kDK) + h * kDK;
        #pragma unroll
        for (int v = 0; v < 8; ++v) {
            const int m = v + halfl * 8;
            const float val = cacc[v] + cpart[1][m][ntile * 16 + n16];
            cb[(size_t)m * (kH * kDK) + ntile * 16 + n16] = (__bf16)val;
        }
    }
}

// ===================== output projection =====================
// Out[r,n] = sum_k Xc[r,k] * Wo[k,n] + bo[n];  64x64 block, 2 WMMAs/wave/K-step
__global__ __launch_bounds__(256)
void out_proj_kernel(const __bf16* __restrict__ Xc,  // [B*S, H*DK] bf16
                     const float*  __restrict__ Wo,  // [H*DK, D]
                     const float*  __restrict__ bo,  // [D]
                     float* __restrict__ Out)        // [B*S, D]
{
    __shared__ __bf16   sA[64][40];
    __shared__ unsigned sBp[16][68];

    const int tid   = threadIdx.x;
    const int lane  = tid & 31;
    const int wave  = tid >> 5;
    const int n16   = lane & 15;
    const int halfl = lane >> 4;

    const int row0 = blockIdx.x * 64;
    const int col0 = blockIdx.y * 64;
    const int mtile = wave >> 1;   // 0..3
    const int nhalf = wave & 1;    // 0..1

    const int ar  = tid >> 3;
    const int aq  = (tid & 7) * 4;
    const int br2 = tid >> 4;
    const int bc4 = (tid & 15) * 4;

    v8f accs[2] = {{}, {}};

    for (int k0 = 0; k0 < kD; k0 += 32) {
        // stage A: already bf16, straight b64 copies
        #pragma unroll
        for (int j = 0; j < 2; ++j) {
            const int r = ar + 32 * j;
            *(v2u*)(&sA[r][aq]) = *(const v2u*)(Xc + (size_t)(row0 + r) * kD + k0 + aq);
        }
        // stage B packed pairs from fp32 Wo
        {
            const float* r0 = Wo + (size_t)(k0 + 2 * br2) * kD + col0 + bc4;
            v4f wa = *(const v4f*)(r0);
            v4f wb = *(const v4f*)(r0 + kD);
            v4u w;
            #pragma unroll
            for (int j = 0; j < 4; ++j) w[j] = pack_bf16_f32(wa[j], wb[j]);
            *(v4u*)(&sBp[br2][bc4]) = w;
        }
        __builtin_prefetch(Xc + (size_t)(row0 + ar) * kD + k0 + 32 + aq, 0, 0);
        __syncthreads();

        Frag a;
        const __bf16* arow = &sA[mtile * 16 + n16][0];
        a.q2[0] = *(const v4u*)(arow + halfl * 16);
        a.q2[1] = *(const v4u*)(arow + halfl * 16 + 8);

        Frag b0, b1;
        #pragma unroll
        for (int v = 0; v < 8; ++v) {
            b0.w[v] = sBp[halfl * 8 + v][(nhalf * 2 + 0) * 16 + n16];
            b1.w[v] = sBp[halfl * 8 + v][(nhalf * 2 + 1) * 16 + n16];
        }
        accs[0] = WMMA_BF16(a.f, b0.f, accs[0]);
        accs[1] = WMMA_BF16(a.f, b1.f, accs[1]);
        __syncthreads();
    }

    #pragma unroll
    for (int t = 0; t < 2; ++t) {
        const int ntile = nhalf * 2 + t;
        const float bval = bo[col0 + ntile * 16 + n16];
        #pragma unroll
        for (int v = 0; v < 8; ++v) {
            const int m = mtile * 16 + v + halfl * 8;
            Out[(size_t)(row0 + m) * kD + col0 + ntile * 16 + n16] = accs[t][v] + bval;
        }
    }
}

// ===================== host launcher =====================
extern "C" void kernel_launch(void* const* d_in, const int* in_sizes, int n_in,
                              void* d_out, int out_size, void* d_ws, size_t ws_size,
                              hipStream_t stream) {
    const float* query = (const float*)d_in[0];
    const float* key_  = (const float*)d_in[1];
    const float* value = (const float*)d_in[2];
    const float* Wq    = (const float*)d_in[3];
    const float* bq    = (const float*)d_in[4];
    const float* Wk    = (const float*)d_in[5];
    const float* bk    = (const float*)d_in[6];
    const float* Wv    = (const float*)d_in[7];
    const float* bv    = (const float*)d_in[8];
    const float* Wo    = (const float*)d_in[9];
    const float* bo    = (const float*)d_in[10];

    float* out  = (float*)d_out;                 // [B,S,D]
    float* attn = out + (size_t)kB * kS * kD;    // [B,H,S,S]

    char* ws = (char*)d_ws;
    const size_t headBytes = (size_t)kB * kH * kS * kDK * sizeof(__bf16);  // 16 MB
    __bf16* qh  = (__bf16*)(ws);
    __bf16* kh  = (__bf16*)(ws + headBytes);
    __bf16* vh  = (__bf16*)(ws + 2 * headBytes);
    __bf16* ctx = (__bf16*)(ws + 3 * headBytes); // [B,S,H*DK]

    dim3 blk(256);

    dim3 gproj(kS / 64, kH, kB);   // 16x16x8 = 2048 blocks
    qkv_proj_kernel<<<gproj, blk, 0, stream>>>(query, Wq, bq, qh);
    qkv_proj_kernel<<<gproj, blk, 0, stream>>>(key_,  Wk, bk, kh);
    qkv_proj_kernel<<<gproj, blk, 0, stream>>>(value, Wv, bv, vh);

    dim3 gattn(kS / 16, kH, kB);   // 64x16x8 = 8192 blocks
    attn_kernel<<<gattn, blk, 0, stream>>>(qh, kh, vh, attn, ctx);

    dim3 gout((kB * kS) / 64, kD / 64, 1);  // 128x16 = 2048 blocks
    out_proj_kernel<<<gout, blk, 0, stream>>>(ctx, Wo, bo, out);
}
